// neuralSDEModel_59261958750958
// MI455X (gfx1250) — compile-verified, matrix operand order checked by reading, not run on hardware
//
#include <hip/hip_runtime.h>
#include <hip/hip_bf16.h>
#include <hip/hip_fp16.h>

typedef __attribute__((ext_vector_type(16))) _Float16 v16h;
typedef __attribute__((ext_vector_type(8)))  float    v8f;
typedef __attribute__((ext_vector_type(4)))  unsigned int v4u;
typedef __attribute__((ext_vector_type(8)))  int      v8i_;
typedef __attribute__((ext_vector_type(4)))  int      v4i_;

#define BB 65536
#define CC 2
#define TT 64
#define HH 64
#define CT 8                 // timestep chunk staged in LDS
#define WAVES 4
#define BLOCK (WAVES * 32)

#define SC_STRIDE (CT + 1)          // 9 words  -> bank-conflict-free (gcd(9,64)=1)
#define VC_STRIDE (3 * CT + 1)      // 25 words -> bank-conflict-free
#define SC_SIZE   (32 * SC_STRIDE)  // per scalar array per wave
#define VC_SIZE   (32 * VC_STRIDE)  // per vec3 array per wave
#define WAVE_LDS  (5 * SC_SIZE + 2 * VC_SIZE + 32)   // 3072 floats = 12 KB / wave

#if defined(__AMDGCN__) && __has_builtin(__builtin_amdgcn_tensor_load_to_lds) && \
    __has_builtin(__builtin_amdgcn_s_wait_tensorcnt)
#define USE_TDM 1
#else
#define USE_TDM 0
#endif

__device__ __forceinline__ unsigned hash_u32(unsigned x) {
    x ^= x >> 16; x *= 0x7feb352du;
    x ^= x >> 15; x *= 0x846ca68bu;
    x ^= x >> 16; return x;
}

// Fast reciprocal / sqrt: single trans-unit instructions, no IEEE expansion.
__device__ __forceinline__ float fast_rcp(float x) {
#if __has_builtin(__builtin_amdgcn_rcpf)
    return __builtin_amdgcn_rcpf(x);
#else
    return 1.0f / x;
#endif
}
__device__ __forceinline__ float fast_sqrt(float x) {
#if __has_builtin(__builtin_amdgcn_sqrtf)
    return __builtin_amdgcn_sqrtf(x);
#else
    return sqrtf(x);
#endif
}

// CDNA5 V_TANH_F32 (confirmed in disasm as a TRANS32 op).
__device__ __forceinline__ float fast_tanh(float x) {
#if __has_builtin(__builtin_amdgcn_tanhf)
    return __builtin_amdgcn_tanhf(x);
#else
    const float t = __expf(2.0f * x);
    return (t - 1.0f) * fast_rcp(t + 1.0f);
#endif
}

// Branch-free softplus: max(o,0) + log(1 + exp(-|o|))
__device__ __forceinline__ float fast_softplus(float o) {
    return fmaxf(o, 0.0f) + __logf(1.0f + __expf(-fabsf(o)));
}

#if USE_TDM
// Issue a TDM 2D tile load: 32 rows x CT DWORDs, row stride CC*TT elements,
// into LDS with pad_amount=1 DWORD every pad_interval=8 DWORDs (== SC_STRIDE 9).
// This toolchain exposes the 6-arg builtin:
//   (uint32x4 g0, int32x8 g1, int32x4 g2, int32x4 g3, int32x8 g4, i32 cpol)
__device__ __forceinline__ void tdm_load_tile(const float* gptr, float* lptr) {
    const unsigned long long ga = (unsigned long long)(uintptr_t)gptr;
    const unsigned ga_lo = __builtin_amdgcn_readfirstlane((unsigned)ga);
    const unsigned ga_hi = __builtin_amdgcn_readfirstlane((unsigned)(ga >> 32));
    const unsigned la    = __builtin_amdgcn_readfirstlane((unsigned)(uintptr_t)lptr);
    // D# group 0: count=1 | lds_addr | global_addr[56:0] | type=2
    v4u g0 = { 1u, la, ga_lo, (ga_hi & 0x01FFFFFFu) | (2u << 30) };
    // D# group 1:
    //  dw0: data_size=2(4B)<<16 | pad_enable<<20 | pad_interval=2(8 DW)<<22 | pad_amount=0(1 DW)<<25
    //  dw1: tensor_dim0(=CT) low16 << 16   (atomic_barrier_addr = 0)
    //  dw2: tensor_dim0 hi16 | tensor_dim1(=32) low16 << 16
    //  dw3: tensor_dim1 hi16 | tile_dim0(=CT) << 16
    //  dw4: tile_dim1(=32) | tile_dim2(=0) << 16
    //  dw5: tensor_dim0_stride = CC*TT (low 32)
    //  dw6,dw7: stride hi / dim1_stride = 0
    v8i_ g1 = { (int)((2u << 16) | (1u << 20) | (2u << 22)),
                (int)((unsigned)CT << 16),
                (int)(32u << 16),
                (int)((unsigned)CT << 16),
                32, (int)(CC * TT), 0, 0 };
    v4i_ gz4 = { 0, 0, 0, 0 };
    v8i_ gz8 = { 0, 0, 0, 0, 0, 0, 0, 0 };
    __builtin_amdgcn_tensor_load_to_lds(g0, g1, gz4, gz4, gz8, 0);
}
#endif

__global__ __launch_bounds__(BLOCK)
void sde_main(const float* __restrict__ step_size, const float* __restrict__ us_n,
              const float* __restrict__ uc,        const float* __restrict__ timescale,
              const float* __restrict__ mesh_ratio,const float* __restrict__ us_np1,
              const float* __restrict__ viscosity, const float* __restrict__ epsilon,
              const float* __restrict__ W1,        const float* __restrict__ b1,
              const float* __restrict__ W2,        const float* __restrict__ b2,
              float* __restrict__ ws)
{
    __shared__ float lds[WAVES * WAVE_LDS];

    const int lane = threadIdx.x & 31;
    const int wave = threadIdx.x >> 5;
    const int bw0  = (blockIdx.x * WAVES + wave) * 32;   // wave's first row
    const int b    = bw0 + lane;

    float* L   = lds + wave * WAVE_LDS;
    float* Ldt = L;
    float* Lts = L + 1 * SC_SIZE;
    float* Lms = L + 2 * SC_SIZE;
    float* Lvi = L + 3 * SC_SIZE;
    float* Lep = L + 4 * SC_SIZE;
    float* Luc = L + 5 * SC_SIZE;
    float* Lgt = L + 5 * SC_SIZE + VC_SIZE;
    float* Lnn = L + 5 * SC_SIZE + 2 * VC_SIZE;

    const int  np = lane & 15;
    const bool lo = lane < 16;

    // Per-lane weight slices (lane np owns column np of each 16-wide N tile).
    _Float16 w1a[4], w1b[4];
    float    w2v[4], b1v[4];
#pragma unroll
    for (int j = 0; j < 4; ++j) {
        w1a[j] = (_Float16)W1[0 * HH + 16 * j + np];
        w1b[j] = (_Float16)W1[1 * HH + 16 * j + np];
        w2v[j] = W2[16 * j + np];
        b1v[j] = b1[16 * j + np];
    }
    const float b2s = b2[0];

    // Row this lane publishes in the folded layer-2 reduction:
    // lanes 0-15 hold row (lane&7), lanes 16-31 hold row 8+(lane&7).
    const int mrow = (lane & 7) | ((lane & 16) >> 1);

    for (int c = 0; c < CC; ++c) {
        const size_t base  = ((size_t)b   * CC + c) * TT;
        const size_t base0 = ((size_t)bw0 * CC + c) * TT;

        float us[3];
        us[0] = us_n[(base + 0) * 3 + 0];
        us[1] = us_n[(base + 0) * 3 + 1];
        us[2] = us_n[(base + 0) * 3 + 2];

        float ini = 0.5f * (us[0]*us[0] + us[1]*us[1] + us[2]*us[2]);
#pragma unroll
        for (int m = 1; m < 32; m <<= 1) ini += __shfl_xor(ini, m, 32);
        if (lane == 0) atomicAdd(&ws[c], ini);

        for (int tc = 0; tc < TT / CT; ++tc) {
            const int t0 = tc * CT;

            // ---- stage chunk into LDS ----
            {
                const float* gs[5] = {step_size, timescale, mesh_ratio, viscosity, epsilon};
                float*       ls[5] = {Ldt, Lts, Lms, Lvi, Lep};
#if USE_TDM
                // Scalar tiles via Tensor Data Mover (DMA, async, LDS-padded).
#pragma unroll
                for (int a = 0; a < 5; ++a)
                    tdm_load_tile(gs[a] + base0 + t0, ls[a]);
#else
#pragma unroll
                for (int a = 0; a < 5; ++a) {
                    const float4* g = (const float4*)(gs[a] + base + t0);
                    float4 v0 = g[0], v1 = g[1];
                    float* d = ls[a] + lane * SC_STRIDE;
                    d[0]=v0.x; d[1]=v0.y; d[2]=v0.z; d[3]=v0.w;
                    d[4]=v1.x; d[5]=v1.y; d[6]=v1.z; d[7]=v1.w;
                }
#endif
                // Vec3 tiles (24-DWORD rows: not a power-of-two pad interval,
                // keep the float4 VMEM path).
                const float4* gu = (const float4*)(uc     + (base + t0) * 3);
                const float4* gg = (const float4*)(us_np1 + (base + t0) * 3);
                float* du = Luc + lane * VC_STRIDE;
                float* dg = Lgt + lane * VC_STRIDE;
#pragma unroll
                for (int q = 0; q < 6; ++q) {
                    float4 a4 = gu[q];
                    du[4*q+0]=a4.x; du[4*q+1]=a4.y; du[4*q+2]=a4.z; du[4*q+3]=a4.w;
                    float4 b4 = gg[q];
                    dg[4*q+0]=b4.x; dg[4*q+1]=b4.y; dg[4*q+2]=b4.z; dg[4*q+3]=b4.w;
                }
                if (tc + 1 < TT / CT) {   // global_prefetch_b8 for the next chunk
                    __builtin_prefetch(uc     + (base + t0 + CT) * 3, 0, 1);
                    __builtin_prefetch(us_np1 + (base + t0 + CT) * 3, 0, 1);
                }
#if USE_TDM
                asm volatile("" ::: "memory");
                __builtin_amdgcn_s_wait_tensorcnt(0);
                asm volatile("" ::: "memory");
#endif
            }

            for (int tt = 0; tt < CT; ++tt) {
                const int t = t0 + tt;
                const float dt    = Ldt[lane * SC_STRIDE + tt];
                const float tsv   = Lts[lane * SC_STRIDE + tt];
                const float msv   = Lms[lane * SC_STRIDE + tt];
                const float viscv = Lvi[lane * SC_STRIDE + tt];
                const float epsv  = Lep[lane * SC_STRIDE + tt];

                // ---- MLP layer 1 via WMMA: D(16x16) = A(16x32,K pad) x W1 + b1 ----
                const float tsx = __shfl_xor(tsv, 16, 32);
                const float msx = __shfl_xor(msv, 16, 32);
#pragma unroll
                for (int rs = 0; rs < 2; ++rs) {
                    const float tr = rs ? tsx : tsv;
                    const float mr = rs ? msx : msv;
                    v16h A = {};
                    A[0] = lo ? (_Float16)tr : (_Float16)0.f;   // K=0
                    A[1] = lo ? (_Float16)mr : (_Float16)0.f;   // K=1 (K=2..31 pad)
                    float p[8] = {0,0,0,0,0,0,0,0};
#pragma unroll
                    for (int j = 0; j < 4; ++j) {
                        v16h Bm = {};
                        Bm[0] = lo ? w1a[j] : (_Float16)0.f;
                        Bm[1] = lo ? w1b[j] : (_Float16)0.f;
                        v8f Cm;
#pragma unroll
                        for (int e = 0; e < 8; ++e) Cm[e] = b1v[j];
                        v8f D = __builtin_amdgcn_wmma_f32_16x16x32_f16(
                                    false, A, false, Bm, (short)0, Cm, false, false);
#pragma unroll
                        for (int e = 0; e < 8; ++e) p[e] += fast_tanh(D[e]) * w2v[j];
                    }
                    // Folded layer-2 reduction: 8 partials x 16 lanes -> 8 row sums
                    // in 8 shuffles (register count halves each xor step).
                    float q[4];
#pragma unroll
                    for (int i = 0; i < 4; ++i) {
                        const float a = p[2*i], bb = p[2*i+1];
                        const float keep = (lane & 1) ? bb : a;
                        const float send = (lane & 1) ? a  : bb;
                        q[i] = keep + __shfl_xor(send, 1, 32);
                    }
                    float s2[2];
#pragma unroll
                    for (int i = 0; i < 2; ++i) {
                        const float a = q[2*i], bb = q[2*i+1];
                        const float keep = (lane & 2) ? bb : a;
                        const float send = (lane & 2) ? a  : bb;
                        s2[i] = keep + __shfl_xor(send, 2, 32);
                    }
                    {
                        const float a = s2[0], bb = s2[1];
                        const float keep = (lane & 4) ? bb : a;
                        const float send = (lane & 4) ? a  : bb;
                        float v = keep + __shfl_xor(send, 4, 32);
                        v += __shfl_xor(v, 8, 32);      // lane holds row (lane&7)(+8 hi half)
                        const float nn = fast_softplus(v + b2s) + 0.001f;
                        if ((lane & 8) == 0) Lnn[rs * 16 + mrow] = nn;
                    }
                }
                const float nn = Lnn[lane];

                // ---- OU-process update ----
                float drift = 0.8f * fast_rcp(nn) * fast_sqrt(viscv) * fast_rcp(tsv);
                drift = (drift == drift) ? drift : 1.0e9f;
                const float dtime = fast_rcp(drift);
                float dif = fast_sqrt(epsv);
                dif = (dif == dif) ? dif : 1e-6f;
                const float g   = dif * dif * dtime * 0.5f * (1.0f - __expf(-2.0f * dt * drift));
                const float P11 = fast_sqrt(g);
                const float edt = __expf(-dt * drift);

                // counter-based Gaussian noise (Box-Muller)
                const unsigned sbase = ((unsigned)(c * TT + t) * 0x9E3779B9u) ^ 0x2545F491u;
                const unsigned k0 = hash_u32(sbase ^ (unsigned)(b * 4 + 0));
                const unsigned k1 = hash_u32(sbase ^ (unsigned)(b * 4 + 1));
                const unsigned k2 = hash_u32(sbase ^ (unsigned)(b * 4 + 2));
                const unsigned k3 = hash_u32(sbase ^ (unsigned)(b * 4 + 3));
                const float u1 = fmaxf((k0 >> 8) * 0x1p-24f, 1e-7f);
                const float u2 = (k1 >> 8) * 0x1p-24f;
                const float u3 = fmaxf((k2 >> 8) * 0x1p-24f, 1e-7f);
                const float u4 = (k3 >> 8) * 0x1p-24f;
                const float r1 = fast_sqrt(-2.f * __logf(u1));
                const float r2 = fast_sqrt(-2.f * __logf(u3));
                float s1, c1, s2c, c2;
                __sincosf(6.28318530718f * u2, &s1, &c1);
                __sincosf(6.28318530718f * u4, &s2c, &c2);
                const float nz[3] = {r1 * c1, r1 * s1, r2 * c2};

                float tn = 0.f, tg = 0.f;
#pragma unroll
                for (int k = 0; k < 3; ++k) {
                    const float u  = Luc[lane * VC_STRIDE + 3 * tt + k];
                    const float gt = Lgt[lane * VC_STRIDE + 3 * tt + k];
                    const float un = us[k] * edt + u * (1.f - edt) + P11 * nz[k];
                    us[k] = un;
                    tn += 0.5f * un * un;
                    tg += 0.5f * gt * gt;
                }
                // Joint reduction: tn on even lanes, tg on odd lanes (5 shuffles).
                {
                    const float keep = (lane & 1) ? tg : tn;
                    const float send = (lane & 1) ? tn : tg;
                    float v = keep + __shfl_xor(send, 1, 32);
                    v += __shfl_xor(v, 2, 32);
                    v += __shfl_xor(v, 4, 32);
                    v += __shfl_xor(v, 8, 32);
                    v += __shfl_xor(v, 16, 32);
                    if (lane == 0) atomicAdd(&ws[2 + CC * TT + c * TT + t], v);  // tn
                    if (lane == 1) atomicAdd(&ws[2 + c * TT + t], v);            // tg
                }
            }
        }
    }
}

__global__ void zero_ws(float* ws, int n) {
    int i = blockIdx.x * blockDim.x + threadIdx.x;
    if (i < n) ws[i] = 0.f;
}

__global__ void finalize(const float* __restrict__ ws, float* __restrict__ out) {
    if (threadIdx.x == 0 && blockIdx.x == 0) {
        float loss = 0.f;
        for (int c = 0; c < CC; ++c) {
            const float inv = 1.0f / ws[c];     // B cancels: |Sgt-Snn|/Sinit
            for (int t = 0; t < TT; ++t) {
                const float gt = ws[2 + c * TT + t];
                const float nn = ws[2 + CC * TT + c * TT + t];
                loss += fabsf(gt - nn) * inv;
            }
        }
        out[0] = loss;
    }
}

extern "C" void kernel_launch(void* const* d_in, const int* in_sizes, int n_in,
                              void* d_out, int out_size, void* d_ws, size_t ws_size,
                              hipStream_t stream) {
    const float* step_size  = (const float*)d_in[0];
    const float* us_n       = (const float*)d_in[1];
    // d_in[2] = up_n (unused by reference)
    const float* uc         = (const float*)d_in[3];
    const float* timescale  = (const float*)d_in[4];
    const float* mesh_ratio = (const float*)d_in[5];
    const float* us_np1     = (const float*)d_in[6];
    // d_in[7] = up_np1 (unused by reference)
    const float* viscosity  = (const float*)d_in[8];
    const float* epsilon    = (const float*)d_in[9];
    const float* W1         = (const float*)d_in[10];
    const float* b1         = (const float*)d_in[11];
    const float* W2         = (const float*)d_in[12];
    const float* b2         = (const float*)d_in[13];
    float* ws = (float*)d_ws;

    const int nws = 2 + 2 * CC * TT;   // 2 init sums + gt/nn per (c,t)
    zero_ws<<<(nws + 255) / 256, 256, 0, stream>>>(ws, nws);
    sde_main<<<BB / (32 * WAVES), BLOCK, 0, stream>>>(
        step_size, us_n, uc, timescale, mesh_ratio, us_np1,
        viscosity, epsilon, W1, b1, W2, b2, ws);
    finalize<<<1, 32, 0, stream>>>(ws, (float*)d_out);
}
